// MyNet_6794638262868
// MI455X (gfx1250) — matrix operand order repaired and checked by reading, not compiled, not run on hardware
//
#include <hip/hip_runtime.h>
#include <hip/hip_bf16.h>
#include <math.h>

typedef __attribute__((ext_vector_type(16))) _Float16 v16h;
typedef __attribute__((ext_vector_type(8)))  _Float16 v8h;
typedef __attribute__((ext_vector_type(8)))  float    v8f;

#define LTOK 2304          // H*W = 48*48
#define CCH  128           // d_model
#define DI   256           // d_inner
#define DS   16            // d_state
#define NSEQ 8             // 4 directions * B
#define NB   2             // batch
#define RTOT (NSEQ*LTOK)   // 18432 rows through the mamba core
#define NCHUNK 32          // scan chunks per sequence
#define CLEN (LTOK/NCHUNK) // 72 steps per chunk

static __device__ __forceinline__ float sigmoidf_(float x){ return 1.f/(1.f+__expf(-x)); }

// ---------------- WMMA fragment loaders (ISA 7.12.2 layouts) ----------------
// A (16xK, f16, row-major): lane m=lane&15, half g=lane>>4.
// elements 0..7  <- K = kbase + g*8 + e
// elements 8..15 <- K = kbase + 16 + g*8 + (e-8)
static __device__ __forceinline__ v16h load_a_f16(const _Float16* base, int ld,
                                                  int row0, int kbase, int lane){
  int r = lane & 15, g = lane >> 4;
  const _Float16* p = base + (size_t)(row0 + r) * ld + kbase + g * 8;
  v8h lo = *(const v8h*)p;
  v8h hi = *(const v8h*)(p + 16);
  v16h f;
#pragma unroll
  for (int i = 0; i < 8; ++i) { f[i] = lo[i]; f[8 + i] = hi[i]; }
  return f;
}
static __device__ __forceinline__ v16h load_a_f32(const float* base, int ld,
                                                  int row0, int kbase, int lane){
  int r = lane & 15, g = lane >> 4;
  const float* p = base + (size_t)(row0 + r) * ld + kbase + g * 8;
  v8f lo = *(const v8f*)p;
  v8f hi = *(const v8f*)(p + 16);
  v16h f;
#pragma unroll
  for (int i = 0; i < 8; ++i) { f[i] = (_Float16)lo[i]; f[8 + i] = (_Float16)hi[i]; }
  return f;
}
// B (Kx16, f16, stored as 16 rows of length K i.e. column n is contiguous in K):
// lane n=lane&15 holds column col0+n; elements 0..7 <- K = kbase + (lane>>4)*8 + e, elements 8..15 <- K = kbase + 16 + (lane>>4)*8 + (e-8)
static __device__ __forceinline__ v16h load_b_f16(const _Float16* base, int ld,
                                                  int col0, int kbase, int lane){
  int n = lane & 15, g = lane >> 4;
  const _Float16* p = base + (size_t)(col0 + n) * ld + kbase + g * 8;
  v8h lo = *(const v8h*)p;
  v8h hi = *(const v8h*)(p + 16);
  v16h f;
#pragma unroll
  for (int i = 0; i < 8; ++i) { f[i] = lo[i]; f[8 + i] = hi[i]; }
  return f;
}

// One wave computes a 16 x (NT*16) strip of D = A*B with K-loop in steps of 32.
template <int NT, bool AF32, typename EPI>
static __device__ __forceinline__ void gemm_strip(const void* A, int lda, int mbase,
                                                  const _Float16* B, int ldb, int nbase,
                                                  int K, int lane, EPI epi){
  v8f acc[NT];
#pragma unroll
  for (int t = 0; t < NT; ++t)
#pragma unroll
    for (int e = 0; e < 8; ++e) acc[t][e] = 0.f;

  for (int k = 0; k < K; k += 32){
    v16h a;
    if constexpr (AF32) a = load_a_f32((const float*)A,    lda, mbase, k, lane);
    else                a = load_a_f16((const _Float16*)A, lda, mbase, k, lane);
#pragma unroll
    for (int t = 0; t < NT; ++t){
      v16h b = load_b_f16(B, ldb, nbase + t * 16, k, lane);
      acc[t] = __builtin_amdgcn_wmma_f32_16x16x32_f16(false, a, false, b,
                                                      (short)0, acc[t], false, false);
    }
  }
  // D layout: VGPR r -> M = r + 8*(lane>>4), N = lane&15
  int m_off = (lane >> 4) * 8;
  int n0 = nbase + (lane & 15);
#pragma unroll
  for (int t = 0; t < NT; ++t)
#pragma unroll
    for (int r = 0; r < 8; ++r)
      epi(mbase + m_off + r, n0 + t * 16, acc[t][r]);
}

// ---------------- prep kernels ----------------
__global__ void k_cvt_f16(const float* __restrict__ src, _Float16* __restrict__ dst, int n){
  int i = blockIdx.x * blockDim.x + threadIdx.x;
  if (i < n) dst[i] = (_Float16)src[i];
}
__global__ void k_prep_xproj(const float* __restrict__ w, _Float16* __restrict__ dst){
  int i = blockIdx.x * blockDim.x + threadIdx.x;       // 48*256, rows 40..47 zero-padded
  if (i >= 48 * 256) return;
  int j = i >> 8, c = i & 255;
  dst[i] = (j < 40) ? (_Float16)w[j * 256 + c] : (_Float16)0.f;
}
__global__ void k_prep_bn(const float* g1, const float* b1, const float* m1, const float* v1,
                          const float* g2, const float* b2, const float* m2, const float* v2,
                          float* s1, float* sh1, float* s2, float* sh2){
  int i = threadIdx.x;
  if (i < CCH){
    float inv1 = g1[i] * rsqrtf(v1[i] + 1e-5f);
    s1[i] = inv1; sh1[i] = b1[i] - m1[i] * inv1;
    float inv2 = g2[i] * rsqrtf(v2[i] + 1e-5f);
    s2[i] = inv2; sh2[i] = b2[i] - m2[i] * inv2;
  }
}
__global__ void k_prep_aneg(const float* __restrict__ alog, float* __restrict__ an){
  int i = blockIdx.x * blockDim.x + threadIdx.x;
  if (i < DI * DS) an[i] = -__expf(alog[i]);
}
__global__ void k_transpose_x(const float* __restrict__ x, _Float16* __restrict__ xt){
  int i = blockIdx.x * blockDim.x + threadIdx.x;       // (b,l,c)
  if (i >= NB * LTOK * CCH) return;
  int c = i & 127; int t = i >> 7; int l = t % LTOK; int b = t / LTOK;
  xt[i] = (_Float16)x[((size_t)b * CCH + c) * LTOK + l];
}

// ---------------- GEMM1: y = relu(bn1(nin_w @ x)); fused 4-direction scatter ----------------
__global__ void k_gemm1(const _Float16* __restrict__ Wn, const _Float16* __restrict__ Xt,
                        const float* __restrict__ s1, const float* __restrict__ sh1,
                        float* __restrict__ act, _Float16* __restrict__ stacked){
  int lane = threadIdx.x;
  int nbase = blockIdx.x * 64;       // pixel l
  int mbase = blockIdx.y * 16;       // out channel o
  int b = blockIdx.z;
  const _Float16* Bm = Xt + (size_t)b * LTOK * CCH;
  gemm_strip<4, false>(Wn, CCH, mbase, Bm, CCH, nbase, CCH, lane,
    [&](int o, int l, float v){
      float y = fmaxf(v * s1[o] + sh1[o], 0.f);
      act[((size_t)b * CCH + o) * LTOK + l] = y;
      _Float16 h = (_Float16)y;
      int oc = CCH - 1 - o, lr = LTOK - 1 - l;
      stacked[((size_t)(0 * NB + b) * LTOK + l ) * CCH + o ] = h;  // identity
      stacked[((size_t)(1 * NB + b) * LTOK + l ) * CCH + oc] = h;  // flip channels
      stacked[((size_t)(2 * NB + b) * LTOK + lr) * CCH + o ] = h;  // flip tokens
      stacked[((size_t)(3 * NB + b) * LTOK + lr) * CCH + oc] = h;  // both
    });
}

// ---------------- GEMM2: xz = stacked @ in_proj_w.T  (18432 x 512) ----------------
__global__ void k_gemm2(const _Float16* __restrict__ Stk, const _Float16* __restrict__ Wip,
                        _Float16* __restrict__ xz){
  int lane = threadIdx.x;
  int nbase = blockIdx.x * 64;       // over 512
  int mbase = blockIdx.y * 16;       // over 18432
  gemm_strip<4, false>(Stk, CCH, mbase, Wip, CCH, nbase, CCH, lane,
    [&](int m, int n, float v){ xz[(size_t)m * 512 + n] = (_Float16)v; });
}

// ---------------- causal depthwise conv (k=4) + SiLU ----------------
__global__ void k_conv_silu(const _Float16* __restrict__ xz, const float* __restrict__ cw,
                            const float* __restrict__ cb, float* __restrict__ xhf){
  int i = blockIdx.x * blockDim.x + threadIdx.x;
  if (i >= RTOT * DI) return;
  int di = i & 255; int r = i >> 8;
  int l = r % LTOK; int n = r / LTOK;
  float acc = cb[di];
#pragma unroll
  for (int j = 0; j < 4; ++j){
    int ll = l - 3 + j;
    if (ll >= 0) acc += cw[di * 4 + j] * (float)xz[((size_t)(n * LTOK + ll)) * 512 + di];
  }
  xhf[i] = acc * sigmoidf_(acc);
}

// ---------------- GEMM3: x_dbl = xh @ x_proj_w.T (N padded 40 -> 48) ----------------
__global__ void k_gemm3(const float* __restrict__ xhf, const _Float16* __restrict__ Wxp,
                        float* __restrict__ xdbl){
  int lane = threadIdx.x;
  int mbase = blockIdx.y * 16;
  gemm_strip<3, true>(xhf, DI, mbase, Wxp, DI, 0, DI, lane,
    [&](int m, int n, float v){ if (n < 40) xdbl[(size_t)m * 40 + n] = v; });
}

// ---------------- dt = softplus(dt_raw @ dt_w.T + dt_b) ----------------
__global__ void k_dt(const float* __restrict__ xdbl, const float* __restrict__ dtw,
                     const float* __restrict__ dtb, float* __restrict__ dt){
  int i = blockIdx.x * blockDim.x + threadIdx.x;
  if (i >= RTOT * DI) return;
  int di = i & 255; int r = i >> 8;
  float s = dtb[di];
#pragma unroll
  for (int q = 0; q < 8; ++q) s += xdbl[(size_t)r * 40 + q] * dtw[di * 8 + q];
  dt[i] = (s > 25.f) ? s : log1pf(__expf(s));
}

// ---------------- chunked selective scan ----------------
// The per-state decay product over a chunk collapses: prod_t exp(dt_t*A_s) = exp(A_s * sum dt).
// Phase 1: per (seq, chunk, di) run the recurrence with h0 = 0; emit chunk-final state + sum(dt).
__global__ void k_scan_chunk(const float* __restrict__ dt, const float* __restrict__ xhf,
                             const float* __restrict__ xdbl, const float* __restrict__ an,
                             float* __restrict__ hloc, float* __restrict__ ssum){
  int tid = blockIdx.x * blockDim.x + threadIdx.x;   // (n, chunk, di), di fastest
  if (tid >= NSEQ * NCHUNK * DI) return;
  int di = tid & 255; int t = tid >> 8; int c = t % NCHUNK; int n = t / NCHUNK;
  float A[DS], h[DS];
#pragma unroll
  for (int s = 0; s < DS; ++s){ A[s] = an[di * DS + s]; h[s] = 0.f; }
  float S = 0.f;
  for (int j = 0; j < CLEN; ++j){
    size_t r = (size_t)n * LTOK + c * CLEN + j;
    float dtv = dt[r * DI + di];
    float xv  = xhf[r * DI + di];
    float dtx = dtv * xv;
    const float* bc = xdbl + r * 40;
    S += dtv;
#pragma unroll
    for (int s = 0; s < DS; ++s)
      h[s] = __expf(dtv * A[s]) * h[s] + dtx * bc[8 + s];
  }
  ssum[tid] = S;
#pragma unroll
  for (int s = 0; s < DS; ++s) hloc[(size_t)tid * DS + s] = h[s];
}
// Phase 2: per (seq, di) walk 32 chunk summaries sequentially, record incoming state per chunk.
__global__ void k_scan_combine(const float* __restrict__ an, const float* __restrict__ hloc,
                               const float* __restrict__ ssum, float* __restrict__ hin){
  int tid = blockIdx.x * blockDim.x + threadIdx.x;   // NSEQ*DI = 2048
  if (tid >= NSEQ * DI) return;
  int di = tid & 255; int n = tid >> 8;
  float A[DS], h[DS];
#pragma unroll
  for (int s = 0; s < DS; ++s){ A[s] = an[di * DS + s]; h[s] = 0.f; }
  for (int c = 0; c < NCHUNK; ++c){
    size_t idx = (size_t)(n * NCHUNK + c) * DI + di;
#pragma unroll
    for (int s = 0; s < DS; ++s) hin[idx * DS + s] = h[s];
    float S = ssum[idx];
#pragma unroll
    for (int s = 0; s < DS; ++s)
      h[s] = __expf(A[s] * S) * h[s] + hloc[idx * DS + s];
  }
}
// Phase 3: rerun each chunk from its known incoming state and emit y.
__global__ void k_scan_emit(const float* __restrict__ dt, const float* __restrict__ xhf,
                            const float* __restrict__ xdbl, const float* __restrict__ an,
                            const float* __restrict__ hin, float* __restrict__ ys){
  int tid = blockIdx.x * blockDim.x + threadIdx.x;   // (n, chunk, di), di fastest
  if (tid >= NSEQ * NCHUNK * DI) return;
  int di = tid & 255; int t = tid >> 8; int c = t % NCHUNK; int n = t / NCHUNK;
  float A[DS], h[DS];
#pragma unroll
  for (int s = 0; s < DS; ++s){
    A[s] = an[di * DS + s];
    h[s] = hin[(size_t)tid * DS + s];
  }
  for (int j = 0; j < CLEN; ++j){
    size_t r = (size_t)n * LTOK + c * CLEN + j;
    float dtv = dt[r * DI + di];
    float xv  = xhf[r * DI + di];
    float dtx = dtv * xv;
    const float* bc = xdbl + r * 40;
    float acc = 0.f;
#pragma unroll
    for (int s = 0; s < DS; ++s){
      h[s] = __expf(dtv * A[s]) * h[s] + dtx * bc[8 + s];
      acc += h[s] * bc[24 + s];
    }
    ys[r * DI + di] = acc;
  }
}

// ---------------- gate: g = (y + xh*D) * silu(z) ----------------
__global__ void k_gate(const float* __restrict__ ys, const float* __restrict__ xhf,
                       const _Float16* __restrict__ xz, const float* __restrict__ Dp,
                       _Float16* __restrict__ gh){
  int i = blockIdx.x * blockDim.x + threadIdx.x;
  if (i >= RTOT * DI) return;
  int di = i & 255; int r = i >> 8;
  float y = ys[i] + xhf[i] * Dp[di];
  float z = (float)xz[(size_t)r * 512 + 256 + di];
  gh[i] = (_Float16)(y * (z * sigmoidf_(z)));
}

// ---------------- GEMM4: ym = g @ out_proj_w.T (18432 x 128) ----------------
__global__ void k_gemm4(const _Float16* __restrict__ gh, const _Float16* __restrict__ Wout,
                        float* __restrict__ ym){
  int lane = threadIdx.x;
  int nbase = blockIdx.x * 64;  // over 128
  int mbase = blockIdx.y * 16;  // over 18432
  gemm_strip<4, false>(gh, DI, mbase, Wout, DI, nbase, DI, lane,
    [&](int m, int n, float v){ ym[(size_t)m * CCH + n] = v; });
}

// ---------------- combine 4 directions + residual, emit f16 (b,l,c) ----------------
__global__ void k_combine(const float* __restrict__ ym, const float* __restrict__ act,
                          _Float16* __restrict__ combT){
  int i = blockIdx.x * blockDim.x + threadIdx.x;
  if (i >= NB * LTOK * CCH) return;
  int c = i & 127; int t = i >> 7; int l = t % LTOK; int b = t / LTOK;
  int lr = LTOK - 1 - l, cr = CCH - 1 - c;
  float v = 0.25f * ( ym[((size_t)(0 * NB + b) * LTOK + l ) * CCH + c ]
                    + ym[((size_t)(1 * NB + b) * LTOK + l ) * CCH + cr]
                    + ym[((size_t)(2 * NB + b) * LTOK + lr) * CCH + c ]
                    + ym[((size_t)(3 * NB + b) * LTOK + lr) * CCH + cr] )
          + act[((size_t)b * CCH + c) * LTOK + l];
  combT[i] = (_Float16)v;
}

// ---------------- GEMM5: out = relu(bn2(nin2_w @ comb)) ----------------
__global__ void k_gemm5(const _Float16* __restrict__ Wn2, const _Float16* __restrict__ combT,
                        const float* __restrict__ s2, const float* __restrict__ sh2,
                        float* __restrict__ out){
  int lane = threadIdx.x;
  int nbase = blockIdx.x * 64;   // pixel
  int mbase = blockIdx.y * 16;   // out channel
  int b = blockIdx.z;
  gemm_strip<4, false>(Wn2, CCH, mbase, combT + (size_t)b * LTOK * CCH, CCH, nbase, CCH, lane,
    [&](int o, int l, float v){
      out[((size_t)b * CCH + o) * LTOK + l] = fmaxf(v * s2[o] + sh2[o], 0.f);
    });
}

extern "C" void kernel_launch(void* const* d_in, const int* in_sizes, int n_in,
                              void* d_out, int out_size, void* d_ws, size_t ws_size,
                              hipStream_t stream){
  const float* x     = (const float*)d_in[0];
  const float* ninw  = (const float*)d_in[1];
  const float* bn1g  = (const float*)d_in[2];
  const float* bn1b  = (const float*)d_in[3];
  const float* bn1m  = (const float*)d_in[4];
  const float* bn1v  = (const float*)d_in[5];
  const float* ipw   = (const float*)d_in[6];
  const float* cw    = (const float*)d_in[7];
  const float* cb    = (const float*)d_in[8];
  const float* xpw   = (const float*)d_in[9];
  const float* dtw   = (const float*)d_in[10];
  const float* dtbv  = (const float*)d_in[11];
  const float* alog  = (const float*)d_in[12];
  const float* Dp    = (const float*)d_in[13];
  const float* opw   = (const float*)d_in[14];
  const float* nin2w = (const float*)d_in[15];
  const float* bn2g  = (const float*)d_in[16];
  const float* bn2b  = (const float*)d_in[17];
  const float* bn2m  = (const float*)d_in[18];
  const float* bn2v  = (const float*)d_in[19];
  float* out = (float*)d_out;

  char* ws = (char*)d_ws;
  size_t off = 0;
  auto carve = [&](size_t bytes) -> char* {
    char* p = ws + off;
    off = (off + bytes + 255) & ~(size_t)255;
    return p;
  };
  _Float16* Wn   = (_Float16*)carve((size_t)CCH * CCH * 2);
  _Float16* Wip  = (_Float16*)carve((size_t)512 * CCH * 2);
  _Float16* Wxp  = (_Float16*)carve((size_t)48 * DI * 2);
  _Float16* Wout = (_Float16*)carve((size_t)CCH * DI * 2);
  _Float16* Wn2  = (_Float16*)carve((size_t)CCH * CCH * 2);
  float* s1  = (float*)carve(CCH * 4);
  float* sh1 = (float*)carve(CCH * 4);
  float* s2  = (float*)carve(CCH * 4);
  float* sh2 = (float*)carve(CCH * 4);
  float* an  = (float*)carve((size_t)DI * DS * 4);
  _Float16* Xt      = (_Float16*)carve((size_t)NB * LTOK * CCH * 2);
  float*    act     = (float*)   carve((size_t)NB * CCH * LTOK * 4);
  _Float16* stacked = (_Float16*)carve((size_t)NSEQ * LTOK * CCH * 2);
  _Float16* xz      = (_Float16*)carve((size_t)RTOT * 512 * 2);
  float*    xhf     = (float*)   carve((size_t)RTOT * DI * 4);
  float*    xdbl    = (float*)   carve((size_t)RTOT * 40 * 4);
  float*    dtb_    = (float*)   carve((size_t)RTOT * DI * 4);
  float*    ysc     = (float*)   carve((size_t)RTOT * DI * 4);
  _Float16* gh      = (_Float16*)carve((size_t)RTOT * DI * 2);
  float*    ym      = (float*)   carve((size_t)RTOT * CCH * 4);
  _Float16* combT   = (_Float16*)carve((size_t)NB * LTOK * CCH * 2);
  float*    hloc    = (float*)   carve((size_t)NSEQ * NCHUNK * DI * DS * 4);
  float*    hin     = (float*)   carve((size_t)NSEQ * NCHUNK * DI * DS * 4);
  float*    ssum    = (float*)   carve((size_t)NSEQ * NCHUNK * DI * 4);

  auto blocks = [](int n){ return dim3((unsigned)((n + 255) / 256)); };

  // ---- prep ----
  k_cvt_f16<<<blocks(CCH * CCH), 256, 0, stream>>>(ninw, Wn, CCH * CCH);
  k_cvt_f16<<<blocks(512 * CCH), 256, 0, stream>>>(ipw, Wip, 512 * CCH);
  k_cvt_f16<<<blocks(CCH * DI), 256, 0, stream>>>(opw, Wout, CCH * DI);
  k_cvt_f16<<<blocks(CCH * CCH), 256, 0, stream>>>(nin2w, Wn2, CCH * CCH);
  k_prep_xproj<<<blocks(48 * DI), 256, 0, stream>>>(xpw, Wxp);
  k_prep_bn<<<1, 128, 0, stream>>>(bn1g, bn1b, bn1m, bn1v, bn2g, bn2b, bn2m, bn2v,
                                   s1, sh1, s2, sh2);
  k_prep_aneg<<<blocks(DI * DS), 256, 0, stream>>>(alog, an);
  k_transpose_x<<<blocks(NB * LTOK * CCH), 256, 0, stream>>>(x, Xt);

  // ---- pipeline ----
  k_gemm1<<<dim3(LTOK / 64, CCH / 16, NB), 32, 0, stream>>>(Wn, Xt, s1, sh1, act, stacked);
  k_gemm2<<<dim3(512 / 64, RTOT / 16), 32, 0, stream>>>(stacked, Wip, xz);
  k_conv_silu<<<blocks(RTOT * DI), 256, 0, stream>>>(xz, cw, cb, xhf);
  k_gemm3<<<dim3(1, RTOT / 16), 32, 0, stream>>>(xhf, Wxp, xdbl);
  k_dt<<<blocks(RTOT * DI), 256, 0, stream>>>(xdbl, dtw, dtbv, dtb_);
  k_scan_chunk<<<blocks(NSEQ * NCHUNK * DI), 256, 0, stream>>>(dtb_, xhf, xdbl, an, hloc, ssum);
  k_scan_combine<<<blocks(NSEQ * DI), 256, 0, stream>>>(an, hloc, ssum, hin);
  k_scan_emit<<<blocks(NSEQ * NCHUNK * DI), 256, 0, stream>>>(dtb_, xhf, xdbl, an, hin, ysc);
  k_gate<<<blocks(RTOT * DI), 256, 0, stream>>>(ysc, xhf, xz, Dp, gh);
  k_gemm4<<<dim3(CCH / 64, RTOT / 16), 32, 0, stream>>>(gh, Wout, ym);
  k_combine<<<blocks(NB * LTOK * CCH), 256, 0, stream>>>(ym, act, combT);
  k_gemm5<<<dim3(LTOK / 64, CCH / 16, NB), 32, 0, stream>>>(Wn2, combT, s2, sh2, out);
}
